// RuleLearner_35278861369430
// MI455X (gfx1250) — matrix operand order repaired and checked by reading, not compiled, not run on hardware
//
#include <hip/hip_runtime.h>
#include <hip/hip_bf16.h>
#include <math.h>

typedef __attribute__((ext_vector_type(16))) _Float16 v16h;
typedef __attribute__((ext_vector_type(8)))  float    v8f;

#define NI     4
#define SL     5
#define NSYMS  9
#define NP     14            // SL + NSYMS
#define BATCH  65536
#define TPB    128           // 4 waves, 32 batch elements per wave

constexpr int k_INVS[NI][SL] = {{1,2,4,3,4},{2,3,4,5,1},{3,2,4,4,7},{4,1,1,2,3}};
constexpr int k_LABELS[NI]   = {2,3,7,1};

__device__ __forceinline__ float sigmoidf_(float x) { return 1.0f / (1.0f + expf(-x)); }

// log(1 - sigmoid(x)) = -softplus(x), numerically stable
__device__ __forceinline__ float neg_softplus(float x) {
    return (x > 0.0f) ? (-x - log1pf(expf(-x))) : (-log1pf(expf(x)));
}

// A[m=(i*5+l), p] = log(1 - sigmoid(unaryp[i,l,p]) * inv_unary_feat[i,l,p])
// feat: p<5 -> (p==l);  p>=5 -> (INVS[i][l] == p-5).  Zero rows/cols for padding.
__device__ __forceinline__ float a_log_elem(const float* __restrict__ up, int m, int p) {
    if (m >= NI * SL || p >= NP) return 0.0f;
    int i = m / SL, l = m - SL * i;
    int feat = (p < SL) ? (p == l) : (k_INVS[i][l] == (p - SL));
    if (!feat) return 0.0f;
    return neg_softplus(up[(i * SL + l) * NP + p]);
}

__global__ void __launch_bounds__(TPB, 1)
rule_learner_kernel(const int*   __restrict__ ids,      // [B,5]
                    const float* __restrict__ unaryp,   // [4,5,14]
                    const float* __restrict__ binaryp,  // [4,5,5,1]
                    const float* __restrict__ outmapp,  // [4,6]
                    float*       __restrict__ out)      // [B,9]
{
    // Per-wave tile of log(uni0): 32 batch elems x 5 q x 20 m floats
    __shared__ __align__(16) float s_u0[4 * 32 * 100];   // 51200 bytes

    const int wave  = threadIdx.x >> 5;
    const int lane  = threadIdx.x & 31;
    const int laneN = lane & 15;
    const int hi    = lane >> 4;                // 0: lanes 0-15, 1: lanes 16-31
    const int blockBase = blockIdx.x * TPB;
    const int waveBase  = blockBase + wave * 32;
    float* ldsW = &s_u0[wave * 3200];

    // ---------------- Phase 1: log-domain GEMM uni0 via WMMA ----------------
    // uni0[b,(i,l),q] = exp( sum_p log(1-c[(i,l),p]) * (1 - bat_u[b,q,p]) ) with
    // binary bat_u: exact product-of-terms rewrite as a shared-weight GEMM.
    // hi/lo f16 split of A keeps ~f32 accuracy (B entries are exact {0,1} in f16).
    v16h Ahi[2], Alo[2];
#pragma unroll
    for (int t = 0; t < 2; t++) {
#pragma unroll
        for (int e = 0; e < 16; e++) {
            // 16-bit A 16x32 layout: low lanes K=0..7,16..23; high lanes +8
            int K = (e < 8) ? (e + 8 * hi) : (e + 8 + 8 * hi);
            float v  = a_log_elem(unaryp, laneN + 16 * t, K);
            _Float16 vh = (_Float16)v;
            Ahi[t][e] = vh;
            Alo[t][e] = (_Float16)(v - (float)vh);
        }
    }

#pragma unroll
    for (int h = 0; h < 2; h++) {
        const int bId = waveBase + 16 * h + laneN;
        int sid[SL];
#pragma unroll
        for (int q = 0; q < SL; q++) sid[q] = ids[bId * SL + q];

#pragma unroll
        for (int q = 0; q < SL; q++) {
            // B[p, n] = 1 - bat_u[b_n, q, p]; 16-bit B 32x16: low lanes K=0..7,16..23; high lanes +8
            v16h Bv;
#pragma unroll
            for (int e = 0; e < 16; e++) {
                int p = (e < 8) ? (e + 8 * hi) : (e + 8 + 8 * hi);
                float val = 0.0f;
                if (p < NP) {
                    float bu = (p < SL) ? ((p == q) ? 1.0f : 0.0f)
                                        : ((sid[q] == (p - SL)) ? 1.0f : 0.0f);
                    val = 1.0f - bu;
                }
                Bv[e] = (_Float16)val;
            }

            // tile 0: rows m = 0..15 (always valid) -> two unconditional b128 stores
            {
                v8f C = {};
                C = __builtin_amdgcn_wmma_f32_16x16x32_f16(false, Alo[0], false, Bv,
                                                           (short)0, C, false, false);
                C = __builtin_amdgcn_wmma_f32_16x16x32_f16(false, Ahi[0], false, Bv,
                                                           (short)0, C, false, false);
                // C layout: VGPR r -> (M = r + 8*hi, N = laneN)
                float* dst = &ldsW[(16 * h + laneN) * 100 + q * 20 + 8 * hi];
                *(float4*)(dst)     = make_float4(C[0], C[1], C[2], C[3]);
                *(float4*)(dst + 4) = make_float4(C[4], C[5], C[6], C[7]);
            }
            // tile 1: rows m = 16..19 valid only in low lane-half (r=0..3, hi=0)
            {
                v8f C = {};
                C = __builtin_amdgcn_wmma_f32_16x16x32_f16(false, Alo[1], false, Bv,
                                                           (short)0, C, false, false);
                C = __builtin_amdgcn_wmma_f32_16x16x32_f16(false, Ahi[1], false, Bv,
                                                           (short)0, C, false, false);
                if (hi == 0) {
                    float* dst = &ldsW[(16 * h + laneN) * 100 + q * 20 + 16];
                    *(float4*)(dst) = make_float4(C[0], C[1], C[2], C[3]);
                }
            }
        }
    }
    __syncthreads();

    // ---------------- Phase 2: per-batch scalar graph ----------------
    const int b  = blockBase + threadIdx.x;
    const int ng = threadIdx.x & 31;
    const float* u0log = &s_u0[wave * 3200 + ng * 100];

    int sid[SL];
#pragma unroll
    for (int q = 0; q < SL; q++) sid[q] = ids[b * SL + q];

    float uni0[NI][SL][SL];
#pragma unroll
    for (int i = 0; i < NI; i++)
#pragma unroll
        for (int l = 0; l < SL; l++)
#pragma unroll
            for (int q = 0; q < SL; q++)
                uni0[i][l][q] = expf(u0log[q * 20 + i * SL + l]);

    float eqf[SL][SL];
#pragma unroll
    for (int q = 0; q < SL; q++)
#pragma unroll
        for (int qp = 0; qp < SL; qp++)
            eqf[q][qp] = (q != qp && sid[q] == sid[qp]) ? 1.0f : 0.0f;

    float iuni[NI], sym[NI][NSYMS];
#pragma unroll
    for (int i = 0; i < NI; i++) {
        // softmax of inv_out_map_p row (6 entries)
        float omp[6], omx = -1e30f;
#pragma unroll
        for (int j = 0; j < 6; j++) { omp[j] = outmapp[i * 6 + j]; omx = fmaxf(omx, omp[j]); }
        float om[6], osum = 0.0f;
#pragma unroll
        for (int j = 0; j < 6; j++) { om[j] = expf(omp[j] - omx); osum += om[j]; }
        float oinv = 1.0f / osum;
#pragma unroll
        for (int j = 0; j < 6; j++) om[j] *= oinv;

        float nodes[NSYMS];
#pragma unroll
        for (int uu = 0; uu < NSYMS; uu++) nodes[uu] = 0.0f;
        float iu = 1.0f;

#pragma unroll
        for (int l = 0; l < SL; l++) {
            float prodreq[SL] = {1.0f, 1.0f, 1.0f, 1.0f, 1.0f};
            // binary conditions: nonzero only where INVS[i][l]==INVS[i][lp], l!=lp (folds at compile time)
#pragma unroll
            for (int lp = 0; lp < SL; lp++) {
                if (lp != l && k_INVS[i][l] == k_INVS[i][lp]) {
                    float sb = sigmoidf_(binaryp[(i * SL + l) * SL + lp]);
#pragma unroll
                    for (int q = 0; q < SL; q++) {
                        float pr = 1.0f;
#pragma unroll
                        for (int qp = 0; qp < SL; qp++)
                            pr *= 1.0f - uni0[i][lp][qp] * eqf[q][qp];
                        // 1 - sb*(1 - support) with support = 1 - pr
                        prodreq[q] *= 1.0f - sb * pr;
                    }
                }
            }
            float u[SL];
#pragma unroll
            for (int q = 0; q < SL; q++) u[q] = uni0[i][l][q] * prodreq[q];

            // node_select = softmax(10*u) over q
            float mx = u[0];
#pragma unroll
            for (int q = 1; q < SL; q++) mx = fmaxf(mx, u[q]);
            float e[SL], se = 0.0f;
#pragma unroll
            for (int q = 0; q < SL; q++) { e[q] = expf(10.0f * (u[q] - mx)); se += e[q]; }
            float sinv = 1.0f / se;

            // edge_outs folded with inv_out_map[l]: scatter into symbol bins
#pragma unroll
            for (int q = 0; q < SL; q++) {
                float v = om[l] * e[q] * sinv;
                int s = sid[q];
#pragma unroll
                for (int uu = 0; uu < NSYMS; uu++)
                    nodes[uu] += (s == uu) ? v : 0.0f;
            }
            // inv_uni accumulation: prod_l (1 - prod_q (1 - u))
            float pq = 1.0f;
#pragma unroll
            for (int q = 0; q < SL; q++) pq *= 1.0f - u[q];
            iu *= 1.0f - pq;
        }
        iuni[i] = iu;
#pragma unroll
        for (int uu = 0; uu < NSYMS; uu++)
            sym[i][uu] = nodes[uu] + om[5] * ((uu == k_LABELS[i]) ? 1.0f : 0.0f);
    }

    // inv_select = softmax(10*inv_uni) over i; blend rule outputs
    float mx2 = iuni[0];
#pragma unroll
    for (int i = 1; i < NI; i++) mx2 = fmaxf(mx2, iuni[i]);
    float es[NI], ss = 0.0f;
#pragma unroll
    for (int i = 0; i < NI; i++) { es[i] = expf(10.0f * (iuni[i] - mx2)); ss += es[i]; }
    float sinv2 = 1.0f / ss;

#pragma unroll
    for (int uu = 0; uu < NSYMS; uu++) {
        float p = 0.0f;
#pragma unroll
        for (int i = 0; i < NI; i++) p += es[i] * sym[i][uu];
        out[b * NSYMS + uu] = p * sinv2;
    }
}

extern "C" void kernel_launch(void* const* d_in, const int* in_sizes, int n_in,
                              void* d_out, int out_size, void* d_ws, size_t ws_size,
                              hipStream_t stream) {
    const int*   ids     = (const int*)d_in[0];     // [65536,5] int32
    const float* unaryp  = (const float*)d_in[1];   // [4,5,14]
    const float* binaryp = (const float*)d_in[2];   // [4,5,5,1]
    const float* outmapp = (const float*)d_in[3];   // [4,6]
    float* out = (float*)d_out;                     // [65536,9]

    dim3 grid(BATCH / TPB), block(TPB);
    rule_learner_kernel<<<grid, block, 0, stream>>>(ids, unaryp, binaryp, outmapp, out);
}
